// GatedDeltaNet_7146825581235
// MI455X (gfx1250) — matrix operand order repaired and checked by reading, not compiled, not run on hardware
//
#include <hip/hip_runtime.h>
#include <hip/hip_bf16.h>

#define Bb    2
#define Tt    2048
#define Dd    1024
#define Hh    16
#define dd    64
#define INNER 1024
#define ROWS  (Bb * Tt)      // 4096
#define KCONV 4

typedef __attribute__((ext_vector_type(16))) __bf16 v16bf;
typedef __attribute__((ext_vector_type(8)))  float  v8f;
typedef __attribute__((ext_vector_type(4)))  int    v4i;

#if defined(__AMDGCN__) && __has_builtin(__builtin_amdgcn_global_load_async_to_lds_b128)
#define HAS_ASYNC_LDS 1
#else
#define HAS_ASYNC_LDS 0
#endif

#define AS1 __attribute__((address_space(1)))
#define AS3 __attribute__((address_space(3)))

__device__ __forceinline__ void wait_async0() {
#if HAS_ASYNC_LDS
#if __has_builtin(__builtin_amdgcn_s_wait_asynccnt)
    __builtin_amdgcn_s_wait_asynccnt(0);
#else
    asm volatile("s_wait_asynccnt 0x0" ::: "memory");
#endif
#endif
}

// copy 16B global -> LDS (async on gfx1250, sync fallback elsewhere)
__device__ __forceinline__ void cp_g2s_16B(const unsigned short* g, unsigned short* s) {
#if HAS_ASYNC_LDS
    __builtin_amdgcn_global_load_async_to_lds_b128(
        (AS1 v4i*)(uintptr_t)g,
        (AS3 v4i*)(unsigned)(uintptr_t)s, 0, 0);
#else
    *(uint4*)s = *(const uint4*)g;
#endif
}

__device__ __forceinline__ unsigned short f2bf(float f) {
    unsigned int u = __float_as_uint(f);
    unsigned int r = u + 0x7FFFu + ((u >> 16) & 1u);
    return (unsigned short)(r >> 16);
}

__device__ __forceinline__ float silu_f(float x) {
    return x / (1.0f + __expf(-x));
}
__device__ __forceinline__ float sigmoid_f(float x) {
    return 1.0f / (1.0f + __expf(-x));
}

// ---------------------------------------------------------------------------
// fp32 -> bf16 conversion (row-major preserved)
__global__ void cvt_bf16_kernel(const float* __restrict__ in,
                                unsigned short* __restrict__ out, int n) {
    int i = blockIdx.x * 256 + threadIdx.x;
    if (i < n) out[i] = f2bf(in[i]);
}

// W (Kdim x Ndim) fp32 row-major  ->  Wt (Ndim x Kdim) bf16 row-major
__global__ void convt_bf16_kernel(const float* __restrict__ W,
                                  unsigned short* __restrict__ Wt,
                                  int Kdim, int Ndim) {
    int idx = blockIdx.x * 256 + threadIdx.x;
    if (idx < Kdim * Ndim) {
        int n = idx / Kdim;
        int k = idx % Kdim;
        Wt[idx] = f2bf(W[(size_t)k * Ndim + n]);
    }
}

// ---------------------------------------------------------------------------
// WMMA bf16 GEMM:  C[M,N] = A[M,K] * Bt[N,K]^T   (A,Bt bf16 row-major, C fp32)
// 256 threads = 8 waves (4 x 2), each wave -> 32x32 (2x2 tiles of 16x16x32)
// block tile = 128(M) x 64(N); K-slabs of 32 double-buffered in LDS via
// async global->LDS copies (ASYNCcnt) overlapped with WMMA.
__global__ void wmma_gemm_kernel(const unsigned short* __restrict__ A,
                                 const unsigned short* __restrict__ Bt,
                                 float* __restrict__ C,
                                 int M, int N, int Kd) {
    __shared__ unsigned short As[2][128 * 32];   // 2 x 8 KB
    __shared__ unsigned short Bs[2][64 * 32];    // 2 x 4 KB

    const int tid  = threadIdx.x;
    const int wave = tid >> 5;
    const int lane = tid & 31;
    const int wm = wave & 3;        // 0..3
    const int wn = wave >> 2;       // 0..1
    const int bm = blockIdx.x * 128;
    const int bn = blockIdx.y * 64;
    const int m0 = wm * 32;         // local within block tile
    const int n0 = wn * 32;
    const int lr = lane & 15;
    const int hi = lane >> 4;       // 0/1 half-wave

    // staging map: thread copies 8 bf16 (16 B); 256 threads cover 64 rows/pass
    const int srow = tid >> 2;            // 0..63
    const int scol = (tid & 3) * 8;       // 0,8,16,24
    const unsigned short* gA0 = A  + (size_t)(bm + srow)      * Kd + scol;
    const unsigned short* gA1 = A  + (size_t)(bm + 64 + srow) * Kd + scol;
    const unsigned short* gB  = Bt + (size_t)(bn + srow)      * Kd + scol;
    unsigned short* dA0[2] = { &As[0][srow * 32 + scol],        &As[1][srow * 32 + scol] };
    unsigned short* dA1[2] = { &As[0][(64 + srow) * 32 + scol], &As[1][(64 + srow) * 32 + scol] };
    unsigned short* dB [2] = { &Bs[0][srow * 32 + scol],        &Bs[1][srow * 32 + scol] };

    union Frag { uint4 q[2]; v16bf v; };
    v8f acc00 = {}, acc01 = {}, acc10 = {}, acc11 = {};

    // prologue: stage slab 0
    cp_g2s_16B(gA0, dA0[0]);
    cp_g2s_16B(gA1, dA1[0]);
    cp_g2s_16B(gB,  dB[0]);

    int cur = 0;
    for (int kk = 0; kk < Kd; kk += 32) {
        wait_async0();          // my async copies done (slab 'cur')
        __syncthreads();        // everyone's copies visible; prev reads of cur^1 done

        if (kk + 32 < Kd) {
            // overlap: stage next slab into the other buffer while computing
            cp_g2s_16B(gA0 + kk + 32, dA0[cur ^ 1]);
            cp_g2s_16B(gA1 + kk + 32, dA1[cur ^ 1]);
            cp_g2s_16B(gB  + kk + 32, dB [cur ^ 1]);
            if (kk + 64 < Kd) {
                __builtin_prefetch(gA0 + kk + 64, 0, 3);
                __builtin_prefetch(gB  + kk + 64, 0, 3);
            }
        }

        // A frag: halves 0-7 -> K=hi*8+0..7, halves 8-15 -> +16
        // B frag: halves 0-7 -> K=hi*8+0..7, halves 8-15 -> +16
        const unsigned short* a0 = &As[cur][(m0 + lr)      * 32 + hi * 8];
        const unsigned short* a1 = &As[cur][(m0 + 16 + lr) * 32 + hi * 8];
        const unsigned short* b0 = &Bs[cur][(n0 + lr)      * 32 + hi * 8];
        const unsigned short* b1 = &Bs[cur][(n0 + 16 + lr) * 32 + hi * 8];

        Frag fa0, fa1, fb0, fb1;
        fa0.q[0] = *(const uint4*)(a0);
        fa0.q[1] = *(const uint4*)(a0 + 16);
        fa1.q[0] = *(const uint4*)(a1);
        fa1.q[1] = *(const uint4*)(a1 + 16);
        fb0.q[0] = *(const uint4*)(b0);
        fb0.q[1] = *(const uint4*)(b0 + 16);
        fb1.q[0] = *(const uint4*)(b1);
        fb1.q[1] = *(const uint4*)(b1 + 16);

        acc00 = __builtin_amdgcn_wmma_f32_16x16x32_bf16(false, fa0.v, false, fb0.v,
                                                        (short)0, acc00, false, false);
        acc01 = __builtin_amdgcn_wmma_f32_16x16x32_bf16(false, fa0.v, false, fb1.v,
                                                        (short)0, acc01, false, false);
        acc10 = __builtin_amdgcn_wmma_f32_16x16x32_bf16(false, fa1.v, false, fb0.v,
                                                        (short)0, acc10, false, false);
        acc11 = __builtin_amdgcn_wmma_f32_16x16x32_bf16(false, fa1.v, false, fb1.v,
                                                        (short)0, acc11, false, false);
        cur ^= 1;
    }

    // C/D layout: vgpr r -> M = r + hi*8 ; N = lr  (within 16x16 tile)
    #pragma unroll
    for (int r = 0; r < 8; ++r) {
        int mA = bm + m0 + r + hi * 8;
        int mB = bm + m0 + 16 + r + hi * 8;
        int nA = bn + n0 + lr;
        int nB = bn + n0 + 16 + lr;
        C[(size_t)mA * N + nA] = acc00[r];
        C[(size_t)mA * N + nB] = acc01[r];
        C[(size_t)mB * N + nA] = acc10[r];
        C[(size_t)mB * N + nB] = acc11[r];
    }
}

// ---------------------------------------------------------------------------
// alpha/beta: sigmoid(x @ Wa + ba), sigmoid(x @ Wb + bb) ; one block per row
__global__ void ab_kernel(const float* __restrict__ x,
                          const float* __restrict__ Wa, const float* __restrict__ ba,
                          const float* __restrict__ Wb, const float* __restrict__ bb,
                          float* __restrict__ alpha, float* __restrict__ beta) {
    int row = blockIdx.x;
    int tid = threadIdx.x;          // 256
    int hh = tid & 15, seg = tid >> 4;
    __shared__ float xs[Dd];
    __shared__ float ra[16][16], rb[16][16];
    for (int i = tid; i < Dd; i += 256) xs[i] = x[(size_t)row * Dd + i];
    __syncthreads();
    float pa = 0.f, pb = 0.f;
    for (int j = 0; j < 64; ++j) {
        int c = seg * 64 + j;
        float xv = xs[c];
        pa += xv * Wa[(size_t)c * Hh + hh];
        pb += xv * Wb[(size_t)c * Hh + hh];
    }
    ra[seg][hh] = pa;
    rb[seg][hh] = pb;
    __syncthreads();
    if (tid < 16) {
        float sa = ba[tid], sb = bb[tid];
        for (int s = 0; s < 16; ++s) { sa += ra[s][tid]; sb += rb[s][tid]; }
        alpha[(size_t)row * Hh + tid] = sigmoid_f(sa);
        beta [(size_t)row * Hh + tid] = sigmoid_f(sb);
    }
}

// ---------------------------------------------------------------------------
// depthwise causal conv(K=4) + SiLU + optional per-head L2 norm
// grid (ROWS, H), block dd=64: one thread per channel within head
__global__ void conv_silu_kernel(const float* __restrict__ in,
                                 const float* __restrict__ w,
                                 const float* __restrict__ bias,
                                 float* __restrict__ out, int do_norm) {
    int bt = blockIdx.x;
    int c  = blockIdx.y * dd + threadIdx.x;
    int t  = bt & (Tt - 1);
    float acc = bias[c];
    #pragma unroll
    for (int k = 0; k < KCONV; ++k) {
        int trel = t - (KCONV - 1) + k;
        float xv = (trel >= 0) ? in[(size_t)(bt - t + trel) * INNER + c] : 0.0f;
        acc += xv * w[(size_t)c * KCONV + k];
    }
    float s = silu_f(acc);
    if (do_norm) {
        __shared__ float red[dd];
        red[threadIdx.x] = s * s;
        __syncthreads();
        for (int st = 32; st > 0; st >>= 1) {
            if (threadIdx.x < st) red[threadIdx.x] += red[threadIdx.x + st];
            __syncthreads();
        }
        float nrm = sqrtf(red[0]);
        s = s / fmaxf(nrm, 1e-12f);
    }
    out[(size_t)bt * INNER + c] = s;
}

// ---------------------------------------------------------------------------
// gated delta recurrence. One block per (b,h) state; 256 threads.
// thread owns row r = tid>>2, 16 cols starting at (tid&3)*16. Update:
//   Sk = S k ;  S = a*(S - b*Sk k^T) + b*v k^T ;  o = S q
__global__ void scan_kernel(const float* __restrict__ q,
                            const float* __restrict__ k,
                            const float* __restrict__ v,
                            const float* __restrict__ alpha,
                            const float* __restrict__ beta,
                            float* __restrict__ o) {
    int bh = blockIdx.x;
    int b  = bh / Hh;
    int h  = bh % Hh;
    int tid = threadIdx.x;
    int r  = tid >> 2;
    int cg = tid & 3;
    int c0 = cg * 16;

    float S[16];
    #pragma unroll
    for (int j = 0; j < 16; ++j) S[j] = 0.0f;

    __shared__ float ks[dd], qs[dd], vs[dd];
    __shared__ float ab[2];

    for (int t = 0; t < Tt; ++t) {
        size_t bt  = (size_t)b * Tt + t;
        size_t off = (bt * Hh + h) * dd;
        if (tid < 64)        ks[tid]        = k[off + tid];
        else if (tid < 128)  qs[tid - 64]   = q[off + tid - 64];
        else if (tid < 192)  vs[tid - 128]  = v[off + tid - 128];
        else if (tid == 192) {
            ab[0] = alpha[bt * Hh + h];
            ab[1] = beta [bt * Hh + h];
        }
        __syncthreads();

        float a  = ab[0];
        float bw = ab[1];

        // Sk[r] = sum_c S[r,c] * k[c]   (4-lane reduction, lanes grouped in wave)
        float p = 0.f;
        #pragma unroll
        for (int j = 0; j < 16; ++j) p += S[j] * ks[c0 + j];
        p += __shfl_xor(p, 1, 32);
        p += __shfl_xor(p, 2, 32);
        float Sk = p;

        float vr = vs[r];
        float op = 0.f;
        #pragma unroll
        for (int j = 0; j < 16; ++j) {
            float kc = ks[c0 + j];
            float sn = a * (S[j] - bw * Sk * kc) + bw * vr * kc;
            S[j] = sn;
            op += sn * qs[c0 + j];
        }
        op += __shfl_xor(op, 1, 32);
        op += __shfl_xor(op, 2, 32);
        if (cg == 0) o[off + r] = op;
        __syncthreads();
    }
}

// ---------------------------------------------------------------------------
// RMS-norm over INNER + (1+gamma) scale + SiLU(gate) multiply, emit bf16
__global__ void rms_gate_kernel(const float* __restrict__ o,
                                const float* __restrict__ gate_pre,
                                const float* __restrict__ gamma,
                                unsigned short* __restrict__ ob) {
    int row = blockIdx.x;
    int tid = threadIdx.x;  // 256
    __shared__ float red[256];
    float vals[4];
    float ss = 0.f;
    #pragma unroll
    for (int i = 0; i < 4; ++i) {
        vals[i] = o[(size_t)row * INNER + tid * 4 + i];
        ss += vals[i] * vals[i];
    }
    red[tid] = ss;
    __syncthreads();
    for (int st = 128; st > 0; st >>= 1) {
        if (tid < st) red[tid] += red[tid + st];
        __syncthreads();
    }
    float inv = 1.0f / sqrtf(red[0] / (float)INNER + 1e-6f);
    #pragma unroll
    for (int i = 0; i < 4; ++i) {
        int c = tid * 4 + i;
        float g = silu_f(gate_pre[(size_t)row * INNER + c]);
        float val = vals[i] * inv * (1.0f + gamma[c]) * g;
        ob[(size_t)row * INNER + c] = f2bf(val);
    }
}

// ---------------------------------------------------------------------------
extern "C" void kernel_launch(void* const* d_in, const int* in_sizes, int n_in,
                              void* d_out, int out_size, void* d_ws, size_t ws_size,
                              hipStream_t stream) {
    const float* x    = (const float*)d_in[0];
    const float* Wq   = (const float*)d_in[1];
    const float* Wk   = (const float*)d_in[2];
    const float* Wv   = (const float*)d_in[3];
    const float* Wa   = (const float*)d_in[4];
    const float* ba   = (const float*)d_in[5];
    const float* Wb   = (const float*)d_in[6];
    const float* bb   = (const float*)d_in[7];
    const float* Wg   = (const float*)d_in[8];
    const float* Wo   = (const float*)d_in[9];
    const float* cq_w = (const float*)d_in[10];
    const float* cq_b = (const float*)d_in[11];
    const float* ck_w = (const float*)d_in[12];
    const float* ck_b = (const float*)d_in[13];
    const float* cv_w = (const float*)d_in[14];
    const float* cv_b = (const float*)d_in[15];
    const float* gamma= (const float*)d_in[16];

    char* ws = (char*)d_ws;
    size_t off = 0;
    auto alloc = [&](size_t bytes) -> void* {
        off = (off + 255) & ~(size_t)255;
        void* p = ws + off;
        off += bytes;
        return p;
    };

    unsigned short* xb   = (unsigned short*)alloc((size_t)ROWS * Dd * 2);
    unsigned short* wqt  = (unsigned short*)alloc((size_t)Dd * INNER * 2);
    unsigned short* wkt  = (unsigned short*)alloc((size_t)Dd * INNER * 2);
    unsigned short* wvt  = (unsigned short*)alloc((size_t)Dd * INNER * 2);
    unsigned short* wgt  = (unsigned short*)alloc((size_t)Dd * INNER * 2);
    unsigned short* wot  = (unsigned short*)alloc((size_t)INNER * Dd * 2);
    float* qpre  = (float*)alloc((size_t)ROWS * INNER * 4);
    float* kpre  = (float*)alloc((size_t)ROWS * INNER * 4);
    float* vpre  = (float*)alloc((size_t)ROWS * INNER * 4);
    float* gpre  = (float*)alloc((size_t)ROWS * INNER * 4);
    float* qn    = (float*)alloc((size_t)ROWS * INNER * 4);
    float* kn    = (float*)alloc((size_t)ROWS * INNER * 4);
    float* vn    = (float*)alloc((size_t)ROWS * INNER * 4);
    float* alphw = (float*)alloc((size_t)ROWS * Hh * 4);
    float* betw  = (float*)alloc((size_t)ROWS * Hh * 4);
    float* osc   = (float*)alloc((size_t)ROWS * INNER * 4);
    unsigned short* ob = (unsigned short*)alloc((size_t)ROWS * INNER * 2);

    // 1. convert x -> bf16, weights -> bf16 transposed
    cvt_bf16_kernel<<<(ROWS * Dd + 255) / 256, 256, 0, stream>>>(x, xb, ROWS * Dd);
    int nw = Dd * INNER;
    convt_bf16_kernel<<<(nw + 255) / 256, 256, 0, stream>>>(Wq, wqt, Dd, INNER);
    convt_bf16_kernel<<<(nw + 255) / 256, 256, 0, stream>>>(Wk, wkt, Dd, INNER);
    convt_bf16_kernel<<<(nw + 255) / 256, 256, 0, stream>>>(Wv, wvt, Dd, INNER);
    convt_bf16_kernel<<<(nw + 255) / 256, 256, 0, stream>>>(Wg, wgt, Dd, INNER);
    convt_bf16_kernel<<<(nw + 255) / 256, 256, 0, stream>>>(Wo, wot, INNER, Dd);

    // 2. projection GEMMs (WMMA bf16, async-LDS double-buffered)
    dim3 gg(ROWS / 128, INNER / 64);
    wmma_gemm_kernel<<<gg, 256, 0, stream>>>(xb, wqt, qpre, ROWS, INNER, Dd);
    wmma_gemm_kernel<<<gg, 256, 0, stream>>>(xb, wkt, kpre, ROWS, INNER, Dd);
    wmma_gemm_kernel<<<gg, 256, 0, stream>>>(xb, wvt, vpre, ROWS, INNER, Dd);
    wmma_gemm_kernel<<<gg, 256, 0, stream>>>(xb, wgt, gpre, ROWS, INNER, Dd);

    // 3. alpha / beta
    ab_kernel<<<ROWS, 256, 0, stream>>>(x, Wa, ba, Wb, bb, alphw, betw);

    // 4. causal depthwise conv + SiLU (+ L2 norm for q,k)
    dim3 cg(ROWS, Hh);
    conv_silu_kernel<<<cg, dd, 0, stream>>>(qpre, cq_w, cq_b, qn, 1);
    conv_silu_kernel<<<cg, dd, 0, stream>>>(kpre, ck_w, ck_b, kn, 1);
    conv_silu_kernel<<<cg, dd, 0, stream>>>(vpre, cv_w, cv_b, vn, 0);

    // 5. gated delta recurrence
    scan_kernel<<<Bb * Hh, 256, 0, stream>>>(qn, kn, vn, alphw, betw, osc);

    // 6. RMS norm + gate -> bf16
    rms_gate_kernel<<<ROWS, 256, 0, stream>>>(osc, gpre, gamma, ob);

    // 7. output projection (WMMA bf16) -> fp32 d_out
    dim3 go(ROWS / 128, Dd / 64);
    wmma_gemm_kernel<<<go, 256, 0, stream>>>(ob, wot, (float*)d_out, ROWS, Dd, INNER);
}
